// SelfAttention_50354196578549
// MI455X (gfx1250) — matrix operand order repaired and checked by reading, not compiled, not run on hardware
//
#include <hip/hip_runtime.h>
#include <hip/hip_bf16.h>
#include <stdint.h>

typedef __attribute__((ext_vector_type(16))) __bf16 v16bf;
typedef __attribute__((ext_vector_type(8)))  float  v8f;

#define K_HEADS 4
#define K_DH    64
#define K_BATCH 8
#define K_T     2048
#define K_D     512
#define K_INNER 256
#define K_N3    768
#define K_M     (K_BATCH * K_T)   /* 16384 */
#define K_SCALE 0.125f
#define K_LOG2E 1.4426950408889634f

__device__ __forceinline__ uint16_t f2bf(float f) {
    uint32_t u = __float_as_uint(f);
    u += 0x7FFFu + ((u >> 16) & 1u);   // round-to-nearest-even
    return (uint16_t)(u >> 16);
}

union FragU { uint32_t u[8]; v16bf v; };

// A-fragment (16 rows x 32 K) of 16-bit data, row-major source.
// Lane layout: lane L holds row L%16; elements e -> K = (L>=16)*16 + e
// => two contiguous 16B loads at K offsets hi*16 and hi*16+8.
__device__ __forceinline__ v16bf load_frag_a(const uint16_t* __restrict__ src,
                                             int row0, int ld, int k0, int lane) {
    int r  = row0 + (lane & 15);
    int hi = (lane >> 4) & 1;
    const uint16_t* p = src + (size_t)r * ld + k0 + hi * 16;
    FragU f;
    *(uint4*)&f.u[0] = *(const uint4*)p;
    *(uint4*)&f.u[4] = *(const uint4*)(p + 8);
    return f.v;
}

// B-fragment (32 K x 16 cols) of 16-bit data, column-major source srcT[n][k].
// Lane L holds column L%16; elements e -> K = k0 + hi*16 + e => 16 contiguous values.
__device__ __forceinline__ v16bf load_frag_b(const uint16_t* __restrict__ srcT,
                                             int n0, int ld, int k0, int lane) {
    int n  = n0 + (lane & 15);
    int hi = (lane >> 4) & 1;
    const uint16_t* p = srcT + (size_t)n * ld + k0 + hi * 16;
    FragU f;
    *(uint4*)&f.u[0] = *(const uint4*)p;
    *(uint4*)&f.u[4] = *(const uint4*)(p + 8);
    return f.v;
}

__device__ __forceinline__ v8f wmma_bf16(v16bf a, v16bf b, v8f c) {
    return __builtin_amdgcn_wmma_f32_16x16x32_bf16(false, a, false, b, (short)0, c, false, false);
}

// ---------------- conversion / transpose kernels ----------------

__global__ void cvt_f32_to_bf16(const float* __restrict__ in, uint16_t* __restrict__ out, int n) {
    int i = blockIdx.x * blockDim.x + threadIdx.x;
    if (i < n) out[i] = f2bf(in[i]);
}

// in: [rows][cols] f32  ->  out: [cols][rows] bf16
__global__ void transpose_to_bf16(const float* __restrict__ in, uint16_t* __restrict__ out,
                                  int rows, int cols) {
    int i = blockIdx.x * blockDim.x + threadIdx.x;
    if (i < rows * cols) {
        int c = i / rows, r = i - c * rows;
        out[(size_t)c * rows + r] = f2bf(in[(size_t)r * cols + c]);
    }
}

// ---------------- QKV projection: qkv = x @ w_qkv, scatter to Q/K/V^T ----------------
// Wave tile: 32 (M) x 64 (N). wT is w_qkv transposed [768][512] bf16.
__global__ __launch_bounds__(256) void qkv_gemm(const uint16_t* __restrict__ xb,
                                                const uint16_t* __restrict__ wT,
                                                uint16_t* __restrict__ Qb,
                                                uint16_t* __restrict__ Kb,
                                                uint16_t* __restrict__ Vtb) {
    int lane = threadIdx.x & 31;
    int wid  = (blockIdx.x * blockDim.x + threadIdx.x) >> 5;
    const int WN = K_N3 / 64;               // 12
    int m0 = (wid / WN) * 32;
    int n0 = (wid % WN) * 64;

    v8f acc[2][4];
#pragma unroll
    for (int mi = 0; mi < 2; ++mi)
#pragma unroll
        for (int ni = 0; ni < 4; ++ni) acc[mi][ni] = (v8f){};

    for (int k0 = 0; k0 < K_D; k0 += 32) {
        v16bf a0 = load_frag_a(xb, m0,      K_D, k0, lane);
        v16bf a1 = load_frag_a(xb, m0 + 16, K_D, k0, lane);
#pragma unroll
        for (int ni = 0; ni < 4; ++ni) {
            v16bf b = load_frag_b(wT, n0 + ni * 16, K_D, k0, lane);
            acc[0][ni] = wmma_bf16(a0, b, acc[0][ni]);
            acc[1][ni] = wmma_bf16(a1, b, acc[1][ni]);
        }
    }

    int hi = (lane >> 4) & 1;
#pragma unroll
    for (int ni = 0; ni < 4; ++ni) {
        int nb  = n0 + ni * 16;             // wave-uniform
        int sec = nb >> 8;                  // 0=Q 1=K 2=V
        int h   = (nb & 255) >> 6;
        int dh  = (nb & 63) + (lane & 15);
        uint16_t* dst = (sec == 0) ? Qb : ((sec == 1) ? Kb : Vtb);
#pragma unroll
        for (int mi = 0; mi < 2; ++mi) {
#pragma unroll
            for (int r = 0; r < 8; ++r) {
                int m  = m0 + mi * 16 + r + hi * 8;
                int bb = m >> 11;
                int tt = m & (K_T - 1);
                size_t addr;
                if (sec == 2)  // V stored transposed: [bh][dh][t]
                    addr = ((size_t)(bb * K_HEADS + h) * K_DH + dh) * K_T + tt;
                else           // Q/K: [bh][t][dh]
                    addr = ((size_t)(bb * K_HEADS + h) * K_T + tt) * K_DH + dh;
                dst[addr] = f2bf(acc[mi][ni][r]);
            }
        }
    }
}

// ---------------- flash attention: one wave per 16-query tile ----------------
// Computes S^T = K @ Q^T so softmax stats are per-lane; O^T = V^T @ P^T.
__global__ __launch_bounds__(128) void attn_kernel(const uint16_t* __restrict__ Qb,
                                                   const uint16_t* __restrict__ Kb,
                                                   const uint16_t* __restrict__ Vtb,
                                                   uint16_t* __restrict__ Ob) {
    int lane = threadIdx.x & 31;
    int wv   = threadIdx.x >> 5;       // 0..3
    int bh   = blockIdx.x >> 5;        // 0..31
    int qb   = blockIdx.x & 31;        // 0..31
    int q0   = qb * 64 + wv * 16;
    int hi   = (lane >> 4) & 1;

    const uint16_t* Qp = Qb  + (size_t)bh * K_T * K_DH;
    const uint16_t* Kp = Kb  + (size_t)bh * K_T * K_DH;
    const uint16_t* Vp = Vtb + (size_t)bh * K_DH * K_T;

    // Q^T as B-fragments: resident for entire key loop (kdim = dh)
    v16bf bq0 = load_frag_b(Qp, q0, K_DH, 0,  lane);
    v16bf bq1 = load_frag_b(Qp, q0, K_DH, 32, lane);

    v8f o[4];
#pragma unroll
    for (int d = 0; d < 4; ++d) o[d] = (v8f){};
    float m = -1e30f, l = 0.0f;

    for (int kt = 0; kt < K_T; kt += 32) {
        // K rows as A-fragments: 2 key sub-tiles x 2 dh halves
        v16bf a00 = load_frag_a(Kp, kt,      K_DH, 0,  lane);
        v16bf a01 = load_frag_a(Kp, kt,      K_DH, 32, lane);
        v16bf a10 = load_frag_a(Kp, kt + 16, K_DH, 0,  lane);
        v16bf a11 = load_frag_a(Kp, kt + 16, K_DH, 32, lane);

        v8f t0 = (v8f){};
        t0 = wmma_bf16(a00, bq0, t0);
        t0 = wmma_bf16(a01, bq1, t0);
        v8f t1 = (v8f){};
        t1 = wmma_bf16(a10, bq0, t1);
        t1 = wmma_bf16(a11, bq1, t1);

        // online softmax (per-query stats live per lane: column = lane%16)
        float s0[8], s1[8];
        float mx = -1e30f;
#pragma unroll
        for (int r = 0; r < 8; ++r) {
            s0[r] = t0[r] * K_SCALE;
            s1[r] = t1[r] * K_SCALE;
            mx = fmaxf(mx, fmaxf(s0[r], s1[r]));
        }
        mx = fmaxf(mx, __shfl_xor(mx, 16));
        float mnew = fmaxf(m, mx);
        float corr = exp2f((m - mnew) * K_LOG2E);
        float p0[8], p1[8], sum = 0.0f;
#pragma unroll
        for (int r = 0; r < 8; ++r) {
            p0[r] = exp2f((s0[r] - mnew) * K_LOG2E);
            p1[r] = exp2f((s1[r] - mnew) * K_LOG2E);
            sum += p0[r] + p1[r];
        }
        sum += __shfl_xor(sum, 16);
        l = l * corr + sum;
        m = mnew;

        // P^T (C-layout) -> B-fragment: half-wave swap only
        float elo[8], ehi[8];
#pragma unroll
        for (int r = 0; r < 8; ++r) {
            float x1 = __shfl_xor(p1[r], 16);
            float x0 = __shfl_xor(p0[r], 16);
            elo[r] = hi ? x1 : p0[r];   // element r    (key = r   | 16+r)
            ehi[r] = hi ? p1[r] : x0;   // element 8+r  (key = 8+r | 24+r)
        }
        FragU bp;
#pragma unroll
        for (int j = 0; j < 4; ++j) {
            bp.u[j]     = (uint32_t)f2bf(elo[2 * j]) | ((uint32_t)f2bf(elo[2 * j + 1]) << 16);
            bp.u[4 + j] = (uint32_t)f2bf(ehi[2 * j]) | ((uint32_t)f2bf(ehi[2 * j + 1]) << 16);
        }

        // O^T += V^T @ P^T  (4 dh row-tiles)
#pragma unroll
        for (int d = 0; d < 4; ++d) {
#pragma unroll
            for (int r = 0; r < 8; ++r) o[d][r] *= corr;
            v16bf av = load_frag_a(Vp, d * 16, K_T, kt, lane);
            o[d] = wmma_bf16(av, bp.v, o[d]);
        }
    }

    float rl = 1.0f / l;
    int bb = bh >> 2, h = bh & 3;
    int q  = q0 + (lane & 15);
    uint16_t* dst = Ob + ((size_t)bb * K_T + q) * K_INNER + h * K_DH;
#pragma unroll
    for (int d = 0; d < 4; ++d)
#pragma unroll
        for (int r = 0; r < 8; ++r)
            dst[d * 16 + r + hi * 8] = f2bf(o[d][r] * rl);
}

// ---------------- output projection: out = attn @ w_out + b_out (f32 out) ----------------
__global__ __launch_bounds__(256) void out_gemm(const uint16_t* __restrict__ ab,
                                                const uint16_t* __restrict__ wT,
                                                const float* __restrict__ bias,
                                                float* __restrict__ out) {
    int lane = threadIdx.x & 31;
    int wid  = (blockIdx.x * blockDim.x + threadIdx.x) >> 5;
    const int WN = K_D / 64;                // 8
    int m0 = (wid / WN) * 32;
    int n0 = (wid % WN) * 64;

    v8f acc[2][4];
#pragma unroll
    for (int mi = 0; mi < 2; ++mi)
#pragma unroll
        for (int ni = 0; ni < 4; ++ni) acc[mi][ni] = (v8f){};

    for (int k0 = 0; k0 < K_INNER; k0 += 32) {
        v16bf a0 = load_frag_a(ab, m0,      K_INNER, k0, lane);
        v16bf a1 = load_frag_a(ab, m0 + 16, K_INNER, k0, lane);
#pragma unroll
        for (int ni = 0; ni < 4; ++ni) {
            v16bf b = load_frag_b(wT, n0 + ni * 16, K_INNER, k0, lane);
            acc[0][ni] = wmma_bf16(a0, b, acc[0][ni]);
            acc[1][ni] = wmma_bf16(a1, b, acc[1][ni]);
        }
    }

    int hi = (lane >> 4) & 1;
#pragma unroll
    for (int ni = 0; ni < 4; ++ni) {
        int n = n0 + ni * 16 + (lane & 15);
        float bv = bias[n];
#pragma unroll
        for (int mi = 0; mi < 2; ++mi)
#pragma unroll
            for (int r = 0; r < 8; ++r) {
                int mrow = m0 + mi * 16 + r + hi * 8;
                out[(size_t)mrow * K_D + n] = acc[mi][ni][r] + bv;
            }
    }
}

// ---------------- host launch ----------------

extern "C" void kernel_launch(void* const* d_in, const int* in_sizes, int n_in,
                              void* d_out, int out_size, void* d_ws, size_t ws_size,
                              hipStream_t stream) {
    const float* x     = (const float*)d_in[0];   // [16384][512]
    const float* w_qkv = (const float*)d_in[1];   // [512][768]
    const float* w_out = (const float*)d_in[2];   // [256][512]
    const float* b_out = (const float*)d_in[3];   // [512]
    float* out = (float*)d_out;

    uint8_t* ws = (uint8_t*)d_ws;
    // workspace layout (bytes)
    const size_t off_xb    = 0;                         // x bf16: 16 MB
    const size_t off_wqkvT = off_xb    + (size_t)K_M * K_D * 2;        // 768 KB
    const size_t off_woutT = off_wqkvT + (size_t)K_N3 * K_D * 2;       // 256 KB
    const size_t off_Q     = off_woutT + (size_t)K_D * K_INNER * 2;    // 8 MB
    const size_t off_K     = off_Q     + (size_t)K_M * K_DH * K_HEADS * 2 / K_HEADS * 1; // see below
    // (Q/K/V^T each hold b*h*t*dh = 8*4*2048*64 = 4M bf16 = 8 MB)
    const size_t seg = (size_t)K_BATCH * K_HEADS * K_T * K_DH * 2;     // 8 MB
    uint16_t* xb    = (uint16_t*)(ws + off_xb);
    uint16_t* wqkvT = (uint16_t*)(ws + off_wqkvT);
    uint16_t* woutT = (uint16_t*)(ws + off_woutT);
    uint16_t* Qb    = (uint16_t*)(ws + off_Q);
    uint16_t* Kb    = (uint16_t*)(ws + off_Q + seg);
    uint16_t* Vtb   = (uint16_t*)(ws + off_Q + 2 * seg);
    uint16_t* Ab    = (uint16_t*)(ws + off_Q + 3 * seg);               // attn out bf16 [16384][256]
    (void)off_K; (void)in_sizes; (void)n_in; (void)out_size; (void)ws_size;

    // 1) convert x to bf16
    {
        int n = K_M * K_D;
        cvt_f32_to_bf16<<<(n + 255) / 256, 256, 0, stream>>>(x, xb, n);
    }
    // 2) transpose weights to column-major bf16 (B-fragment friendly)
    {
        int n = K_D * K_N3;
        transpose_to_bf16<<<(n + 255) / 256, 256, 0, stream>>>(w_qkv, wqkvT, K_D, K_N3);
    }
    {
        int n = K_INNER * K_D;
        transpose_to_bf16<<<(n + 255) / 256, 256, 0, stream>>>(w_out, woutT, K_INNER, K_D);
    }
    // 3) QKV projection (scatter into Q, K, V^T head-major buffers)
    {
        int waves = (K_M / 32) * (K_N3 / 64);   // 6144
        qkv_gemm<<<waves * 32 / 256, 256, 0, stream>>>(xb, wqkvT, Qb, Kb, Vtb);
    }
    // 4) flash attention: 32 (b,h) x 32 q-blocks, 4 waves/block
    {
        attn_kernel<<<32 * 32, 128, 0, stream>>>(Qb, Kb, Vtb, Ab);
    }
    // 5) output projection + bias
    {
        int waves = (K_M / 32) * (K_D / 64);    // 4096
        out_gemm<<<waves * 32 / 256, 256, 0, stream>>>(Ab, woutT, b_out, out);
    }
}
